// BasicBlock_11493332484921
// MI455X (gfx1250) — matrix operand order repaired and checked by reading, not compiled, not run on hardware
//
#include <hip/hip_runtime.h>

// ---------------------------------------------------------------------------
// BasicBlock (ShiftAddNet style):
//   out = relu( BN2( adder2( shiftconv2( q(relu(BN1( adder1( shiftconv1(q(x)) )))) ) ) ) + x )
// shift convs  -> WMMA f16 implicit GEMM (f32 accum), batched B loads
// adder convs  -> VALU |x-w| register-tiled, fused per-channel BN statistics,
//                 async global->LDS weight staging (gfx1250, ASYNCcnt path)
// ---------------------------------------------------------------------------

typedef __attribute__((ext_vector_type(16))) _Float16 v16h;
typedef __attribute__((ext_vector_type(8)))  float    v8f;
typedef __attribute__((ext_vector_type(4)))  int      v4i;
// pointee-address-space typedefs for the async global->LDS builtin
typedef __attribute__((address_space(1))) v4i* as1_v4i_ptr;
typedef __attribute__((address_space(3))) v4i* as3_v4i_ptr;

#define CH   128        // Ci == Co == 128
#define HW   32         // H == W == 32
#define KTAP 9
#define QSCALE 32768.0f // 2^(16-1)

// ---------------------------------------------------------------------------
// Pack shift-conv weights [Co,Ci,3,3] f32 -> [co][tap][ci] f16
__global__ __launch_bounds__(256) void pack_shift(const float* __restrict__ w,
                                                  _Float16* __restrict__ wp) {
  int i = blockIdx.x * 256 + threadIdx.x;           // i = (co*9 + tap)*128 + ci
  if (i >= CH * CH * KTAP) return;
  int co = i / (CH * KTAP);
  int tap = (i / CH) % KTAP;
  int ci = i % CH;
  wp[i] = (_Float16)w[((size_t)co * CH + ci) * KTAP + tap];
}

// Pack adder-conv weights [Co,Ci,3,3] f32 -> [ci][tap][co] f32
__global__ __launch_bounds__(256) void pack_adder(const float* __restrict__ w,
                                                  float* __restrict__ wp) {
  int i = blockIdx.x * 256 + threadIdx.x;           // i = (ci*9 + tap)*128 + co
  if (i >= CH * CH * KTAP) return;
  int ci = i / (CH * KTAP);
  int tap = (i / CH) % KTAP;
  int co = i % CH;
  wp[i] = w[((size_t)co * CH + ci) * KTAP + tap];
}

__global__ void zero_stats(float* s) { s[threadIdx.x] = 0.0f; }

// ---------------------------------------------------------------------------
// Quantize (+ optional BN+ReLU from batch stats) and transpose NCHW f32 ->
// NHWC f16 via LDS tile so both global sides stay coalesced.
// grid: Nimg*32*4 blocks (n, y, 32-channel chunk), 256 threads.
__global__ __launch_bounds__(256) void transform_nhwc(
    const float* __restrict__ src, const float* __restrict__ gamma,
    const float* __restrict__ beta, const float* __restrict__ ssum,
    const float* __restrict__ sss, _Float16* __restrict__ dst,
    float inv_count, int do_bn) {
  __shared__ float tile[32][33];
  int b = blockIdx.x;
  int cchunk = b & 3;
  int y = (b >> 2) & 31;
  int n = b >> 7;
  int t = threadIdx.x;

  for (int i = t; i < 1024; i += 256) {             // read NCHW coalesced
    int c = i >> 5, x = i & 31;
    int cc = cchunk * 32 + c;
    float v = src[(((size_t)n * CH + cc) * HW + y) * HW + x];
    if (do_bn) {
      float m = ssum[cc] * inv_count;
      float var = sss[cc] * inv_count - m * m;
      v = (v - m) * rsqrtf(var + 1e-5f) * gamma[cc] + beta[cc];
      v = fmaxf(v, 0.0f);
    }
    v = rintf(v * QSCALE) * (1.0f / QSCALE);        // round_fixed (RNE)
    tile[c][x] = v;
  }
  __syncthreads();
  for (int i = t; i < 1024; i += 256) {             // write NHWC coalesced
    int x = i >> 5, c = i & 31;
    dst[(((size_t)n * HW + y) * HW + x) * CH + cchunk * 32 + c] =
        (_Float16)tile[c][x];
  }
}

// ---------------------------------------------------------------------------
// WMMA implicit-GEMM 3x3 conv, pad=1.  Xh: NHWC f16, Wp: [co][tap][ci] f16,
// Y: NCHW f32.  One wave computes a 16-pixel x 128-co tile, K=1152 in regs.
// All 8 B fragments are fetched before the WMMA burst so loads batch into one
// clause with staggered waits instead of load->wait->wmma serialization.
__global__ __launch_bounds__(128) void wmma_conv3x3(
    const _Float16* __restrict__ Xh, const _Float16* __restrict__ Wp,
    float* __restrict__ Y) {
  const int lane = threadIdx.x & 31;
  const int wave = threadIdx.x >> 5;
  const int tile = blockIdx.x * 4 + wave;           // 64 tiles per image
  const int n  = tile >> 6;
  const int ti = tile & 63;
  const int y  = ti >> 1;
  const int x0 = (ti & 1) << 4;

  const int m   = lane & 15;                        // A row (pixel), D col id
  const int grp = lane >> 4;
  const int k0  = grp ? 8 : 0;                      // A lane-group K chunks
  const int k1  = grp ? 24 : 16;
  const int col = lane & 15;                        // B lane column

  v8f acc[8] = {};                                  // 8 co-tiles of 16

#pragma unroll
  for (int tap = 0; tap < KTAP; ++tap) {
    const int dy = tap / 3 - 1, dx = tap % 3 - 1;
    const int yy = y + dy;
    const int xm = x0 + m + dx;
    const bool ok = ((unsigned)yy < (unsigned)HW) && ((unsigned)xm < (unsigned)HW);
    const _Float16* abase =
        Xh + ((((size_t)n * HW + (ok ? yy : 0)) * HW + (ok ? xm : 0)) * CH);
    // one base for all 8 B fragments of this tap; per-cb offsets are constant
    const _Float16* bbase =
        Wp + (((size_t)col * KTAP + tap) * CH + grp * 8);

    for (int c0 = 0; c0 < CH; c0 += 32) {
      union { v16h v; int4 q[2]; } A;
      A.q[0] = ok ? *(const int4*)(abase + c0 + k0) : make_int4(0, 0, 0, 0);
      A.q[1] = ok ? *(const int4*)(abase + c0 + k1) : make_int4(0, 0, 0, 0);

      union { v16h v; int4 q[2]; } Bf[8];
#pragma unroll
      for (int cb = 0; cb < 8; ++cb) {              // 16 loads in flight
        const _Float16* bp = bbase + (size_t)cb * 16 * KTAP * CH + c0;
        Bf[cb].q[0] = *(const int4*)(bp);
        Bf[cb].q[1] = *(const int4*)(bp + 16);
      }
#pragma unroll
      for (int cb = 0; cb < 8; ++cb)                // WMMA burst
        acc[cb] = __builtin_amdgcn_wmma_f32_16x16x32_f16(
            false, A.v, false, Bf[cb].v, (short)0, acc[cb], false, false);
    }
  }

  // D layout: VGPR r -> pixel m = r + 8*grp, lane -> co = cb*16 + (lane&15)
#pragma unroll
  for (int cb = 0; cb < 8; ++cb) {
    const int co = cb * 16 + col;
    float* out = Y + ((((size_t)n * CH + co) * HW + y) * HW) + x0 + grp * 8;
    float4 lo = make_float4(acc[cb][0], acc[cb][1], acc[cb][2], acc[cb][3]);
    float4 hi = make_float4(acc[cb][4], acc[cb][5], acc[cb][6], acc[cb][7]);
    *(float4*)(out)     = lo;
    *(float4*)(out + 4) = hi;
  }
}

// ---------------------------------------------------------------------------
// AdderNet conv:  z[n,co,y,x] = -sum_{ci,tap} |X[n,ci,y+dy,x+dx] - w[co,ci,tap]|
// Block tile: full 128 co x one row (32 px).  Thread: 4 co x 4 px.
// Also accumulates per-channel sum / sumsq (batch-stat BN) via f32 atomics.
__global__ __launch_bounds__(256) void adder_conv3x3(
    const float* __restrict__ X, const float* __restrict__ Wp,  // [ci][tap][co]
    float* __restrict__ Z, float* __restrict__ stat_sum,
    float* __restrict__ stat_ss) {
  const int n = blockIdx.x >> 5;
  const int y = blockIdx.x & 31;
  const int t = threadIdx.x;
  const int pxg = t & 7;                            // pixels x = pxg*4 .. +3
  const int cog = t >> 3;                           // co   = cog*4 .. +3

  __shared__ float sx[4][3][36];                    // [ci][row][x+1], pad
  __shared__ float sw[4][KTAP][CH];                 // [ci][tap][co]
  __shared__ float red[2][CH][8];

  float acc[4][4] = {};

  for (int c0 = 0; c0 < CH; c0 += 4) {
    if (c0 + 4 < CH && t == 0)                      // gfx1250 global_prefetch
      __builtin_prefetch(&X[(((size_t)n * CH + c0 + 4) * HW + y) * HW], 0, 1);

    // stage 4 ci x 9 tap x 128 co weights (pre-packed, fully contiguous):
    // async DMA straight into LDS when available (ASYNCcnt path).
#if defined(__gfx1250__) && \
    __has_builtin(__builtin_amdgcn_global_load_async_to_lds_b128)
    {
      const float* gsrc = Wp + (size_t)c0 * KTAP * CH;
      float* lds = (float*)sw;
      for (int i = t; i < (4 * KTAP * CH) / 4; i += 256) {
        __builtin_amdgcn_global_load_async_to_lds_b128(
            (as1_v4i_ptr)(gsrc + i * 4), (as3_v4i_ptr)(lds + i * 4), 0, 0);
      }
#if __has_builtin(__builtin_amdgcn_s_wait_asynccnt)
      __builtin_amdgcn_s_wait_asynccnt(0);
#else
      asm volatile("s_wait_asynccnt 0" ::: "memory");
#endif
    }
#else
    for (int i = t; i < 4 * KTAP * CH; i += 256)
      ((float*)sw)[i] = Wp[(size_t)c0 * KTAP * CH + i];
#endif

    // stage 4 ci x 3 rows x 34 cols of input (zero-padded)
    for (int i = t; i < 4 * 3 * 34; i += 256) {
      int ci = i / 102, rr = (i % 102) / 34, xx = (i % 34) - 1;
      int yy = y + rr - 1;
      float v = 0.0f;
      if ((unsigned)yy < (unsigned)HW && (unsigned)xx < (unsigned)HW)
        v = X[(((size_t)n * CH + c0 + ci) * HW + yy) * HW + xx];
      sx[ci][rr][xx + 1] = v;
    }
    __syncthreads();

#pragma unroll
    for (int ci = 0; ci < 4; ++ci) {
      float xr[3][6];
#pragma unroll
      for (int r = 0; r < 3; ++r)
#pragma unroll
        for (int j = 0; j < 6; ++j) xr[r][j] = sx[ci][r][pxg * 4 + j];
#pragma unroll
      for (int tap = 0; tap < KTAP; ++tap) {
        const int dy = tap / 3, dx = tap % 3;
        float4 wv4 = *(const float4*)&sw[ci][tap][cog * 4];  // ds_load_b128
        const float wq[4] = {wv4.x, wv4.y, wv4.z, wv4.w};
#pragma unroll
        for (int q = 0; q < 4; ++q)
#pragma unroll
          for (int p = 0; p < 4; ++p)
            acc[q][p] += fabsf(xr[dy][p + dx] - wq[q]);      // 2 VALU/tap
      }
    }
    __syncthreads();
  }

  // write z = -acc (float4) and fold BN partials
#pragma unroll
  for (int q = 0; q < 4; ++q) {
    const int co = cog * 4 + q;
    float z0 = -acc[q][0], z1 = -acc[q][1], z2 = -acc[q][2], z3 = -acc[q][3];
    float* out = Z + (((size_t)n * CH + co) * HW + y) * HW + pxg * 4;
    *(float4*)out = make_float4(z0, z1, z2, z3);
    red[0][co][pxg] = z0 + z1 + z2 + z3;
    red[1][co][pxg] = z0 * z0 + z1 * z1 + z2 * z2 + z3 * z3;
  }
  __syncthreads();
  if (t < CH) {
    float rs = 0.0f, rss = 0.0f;
#pragma unroll
    for (int p = 0; p < 8; ++p) { rs += red[0][t][p]; rss += red[1][t][p]; }
    atomicAdd(&stat_sum[t], rs);                    // global_atomic_add_f32
    atomicAdd(&stat_ss[t], rss);
  }
}

// ---------------------------------------------------------------------------
// Final: out = relu( BN2(z) + residual ), elementwise NCHW, float4.
__global__ __launch_bounds__(256) void bn_add_relu(
    const float* __restrict__ Z, const float* __restrict__ Xres,
    const float* __restrict__ gamma, const float* __restrict__ beta,
    const float* __restrict__ ssum, const float* __restrict__ sss,
    float* __restrict__ out, float inv_count, size_t total) {
  size_t e0 = ((size_t)blockIdx.x * 256 + threadIdx.x) * 4;
  if (e0 >= total) return;
  int c = (int)((e0 >> 10) & (CH - 1));             // H*W = 1024
  float m = ssum[c] * inv_count;
  float var = sss[c] * inv_count - m * m;
  float sc = rsqrtf(var + 1e-5f) * gamma[c];
  float sh = beta[c] - m * sc;
  float4 z = *(const float4*)(Z + e0);
  float4 r = *(const float4*)(Xres + e0);
  float4 o;
  o.x = fmaxf(z.x * sc + sh + r.x, 0.0f);
  o.y = fmaxf(z.y * sc + sh + r.y, 0.0f);
  o.z = fmaxf(z.z * sc + sh + r.z, 0.0f);
  o.w = fmaxf(z.w * sc + sh + r.w, 0.0f);
  *(float4*)(out + e0) = o;
}

// ---------------------------------------------------------------------------
extern "C" void kernel_launch(void* const* d_in, const int* in_sizes, int n_in,
                              void* d_out, int out_size, void* d_ws,
                              size_t ws_size, hipStream_t stream) {
  const float* x   = (const float*)d_in[0];
  const float* ws1 = (const float*)d_in[1];
  const float* wa1 = (const float*)d_in[2];
  const float* g1  = (const float*)d_in[3];
  const float* b1  = (const float*)d_in[4];
  const float* ws2 = (const float*)d_in[5];
  const float* wa2 = (const float*)d_in[6];
  const float* g2  = (const float*)d_in[7];
  const float* b2  = (const float*)d_in[8];
  float* out = (float*)d_out;

  const int Nimg = in_sizes[0] / (CH * HW * HW);
  const size_t nelem = (size_t)Nimg * CH * HW * HW;
  const float inv_count = 1.0f / (float)(Nimg * HW * HW);

  char* p = (char*)d_ws;
  auto carve = [&](size_t bytes) {
    void* r = (void*)p;
    p += (bytes + 255) & ~(size_t)255;
    return r;
  };
  _Float16* X1h = (_Float16*)carve(nelem * 2);
  _Float16* X2h = (_Float16*)carve(nelem * 2);
  _Float16* Wp1 = (_Float16*)carve((size_t)CH * CH * KTAP * 2);
  _Float16* Wp2 = (_Float16*)carve((size_t)CH * CH * KTAP * 2);
  float*    Wa1 = (float*)carve((size_t)CH * CH * KTAP * 4);
  float*    Wa2 = (float*)carve((size_t)CH * CH * KTAP * 4);
  float*    Y   = (float*)carve(nelem * 4);
  float*    Zb  = (float*)carve(nelem * 4);
  float*    st  = (float*)carve(512 * 4);           // sum1,ss1,sum2,ss2
  float* sum1 = st;       float* ss1 = st + 128;
  float* sum2 = st + 256; float* ss2 = st + 384;

  const int wgrid  = (CH * CH * KTAP + 255) / 256;  // weight pack grids
  const int convB  = Nimg * 16;                     // 4 waves/block, 64 tiles/img
  const int addB   = Nimg * HW;                     // (n, y)
  const int xfB    = Nimg * HW * 4;                 // (n, y, cchunk)
  const int finB   = (int)((nelem / 4 + 255) / 256);

  pack_shift<<<wgrid, 256, 0, stream>>>(ws1, Wp1);
  pack_shift<<<wgrid, 256, 0, stream>>>(ws2, Wp2);
  pack_adder<<<wgrid, 256, 0, stream>>>(wa1, Wa1);
  pack_adder<<<wgrid, 256, 0, stream>>>(wa2, Wa2);
  zero_stats<<<1, 512, 0, stream>>>(st);

  // stage 1: q(x) -> f16 NHWC ; shift conv 1 ; adder conv 1 (+ BN1 stats)
  transform_nhwc<<<xfB, 256, 0, stream>>>(x, nullptr, nullptr, nullptr, nullptr,
                                          X1h, inv_count, 0);
  wmma_conv3x3<<<convB, 128, 0, stream>>>(X1h, Wp1, Y);
  adder_conv3x3<<<addB, 256, 0, stream>>>(Y, Wa1, Zb, sum1, ss1);

  // stage 2: BN1+relu+q -> f16 NHWC ; shift conv 2 ; adder conv 2 (+BN2 stats)
  transform_nhwc<<<xfB, 256, 0, stream>>>(Zb, g1, b1, sum1, ss1, X2h, inv_count, 1);
  wmma_conv3x3<<<convB, 128, 0, stream>>>(X2h, Wp2, Y);
  adder_conv3x3<<<addB, 256, 0, stream>>>(Y, Wa2, Zb, sum2, ss2);

  // epilogue: BN2 + residual + relu
  bn_add_relu<<<finB, 256, 0, stream>>>(Zb, x, g2, b2, sum2, ss2, out,
                                        inv_count, nelem);
}
